// ModifiedViTSelfAttention_90933047591130
// MI455X (gfx1250) — matrix operand order repaired and checked by reading, not compiled, not run on hardware
//
#include <hip/hip_runtime.h>

#define NUMH 12
#define HD   64
#define HID  768
#define BB   32
#define SS   577
#define KEEPN 289
#define SPAD 608   // 577 padded to 19*32 for the P@V K-loop

typedef __bf16 v16bf __attribute__((ext_vector_type(16)));
typedef __bf16 v8bf  __attribute__((ext_vector_type(8)));
typedef float  v8f   __attribute__((ext_vector_type(8)));

union AFrag { v16bf v; v8bf h[2]; };

static __device__ __forceinline__ unsigned short f2bf(float f) {
  unsigned int u = __builtin_bit_cast(unsigned int, f);
  u += 0x7FFFu + ((u >> 16) & 1u);           // round-to-nearest-even
  return (unsigned short)(u >> 16);
}
static __device__ __forceinline__ int imin(int a, int b) { return a < b ? a : b; }

// ---------------------------------------------------------------- convert
__global__ void cvt_bf16_kernel(const float* __restrict__ src,
                                unsigned short* __restrict__ dst, int n) {
  for (int i = blockIdx.x * blockDim.x + threadIdx.x; i < n;
       i += gridDim.x * blockDim.x)
    dst[i] = f2bf(src[i]);
}

// ---------------------------------------------------------------- projection
// y[m, n] = sum_e X[row(m), e] * W[n, e] + bias[n]   (einsum 'bse,fe->bsf')
// mode 0: q, M=BB*KEEPN, rows gathered via idx, out (BH, KEEPN, 64) bf16
// mode 1: k, M=BB*SS,                         out (BH, SS, 64)    bf16
// mode 2: v, M=BB*SS,                         out (BH, 64, SPAD)  bf16 (transposed)
__global__ __launch_bounds__(256)
void proj_kernel(const unsigned short* __restrict__ Xb,
                 const unsigned short* __restrict__ Wb,
                 const float* __restrict__ bias,
                 const int* __restrict__ idx,
                 unsigned short* __restrict__ out,
                 int mode, int Mtot)
{
  const int lane = threadIdx.x & 31;
  const int wave = threadIdx.x >> 5;
  const int wm   = wave & 3;            // 4 waves along M (16 rows each)
  const int wn   = wave >> 2;           // 2 waves along N (32 cols each)
  const int half = lane >> 4;
  const int nn   = lane & 15;
  const int mBase = blockIdx.x * 64 + wm * 16;
  const int h     = blockIdx.y;         // head == 64-wide N tile
  const int n0    = h * 64 + wn * 32 + nn;
  const int n1    = n0 + 16;

  // A-matrix row for this lane (lanes 0-15 and 16-31 both carry rows m=0..15)
  int mA = imin(mBase + nn, Mtot - 1);
  long rowX;
  if (mode == 0) { int b = mA / KEEPN, i = mA % KEEPN; rowX = (long)b * SS + idx[i]; }
  else            rowX = mA;
  const unsigned short* xrow  = Xb + rowX * (long)HID;
  const unsigned short* wrow0 = Wb + (long)n0 * HID + half * 16;
  const unsigned short* wrow1 = Wb + (long)n1 * HID + half * 16;
  const int off = half * 16;

  v8f c0 = {}, c1 = {};
  for (int kk = 0; kk < HID; kk += 32) {
    __builtin_prefetch(xrow + kk + 128, 0, 1);
    AFrag a;
    a.h[0] = *(const v8bf*)(xrow + kk + off);
    a.h[1] = *(const v8bf*)(xrow + kk + off + 8);
    v16bf b0 = *(const v16bf*)(wrow0 + kk);
    v16bf b1 = *(const v16bf*)(wrow1 + kk);
    c0 = __builtin_amdgcn_wmma_f32_16x16x32_bf16(false, a.v, false, b0, (short)0, c0, false, false);
    c1 = __builtin_amdgcn_wmma_f32_16x16x32_bf16(false, a.v, false, b1, (short)0, c1, false, false);
  }
  const float bias0 = bias[n0], bias1 = bias[n1];
  const int d0 = wn * 32 + nn;

#pragma unroll
  for (int r = 0; r < 8; ++r) {
    int mOut = mBase + r + half * 8;
    if (mOut >= Mtot) continue;
    float y0 = c0[r] + bias0;
    float y1 = c1[r] + bias1;
    if (mode == 0) {
      int b = mOut / KEEPN, i = mOut % KEEPN;
      unsigned short* q = out + ((long)(b * NUMH + h) * KEEPN + i) * HD;
      q[d0]      = f2bf(y0);
      q[d0 + 16] = f2bf(y1);
    } else if (mode == 1) {
      int b = mOut / SS, s = mOut % SS;
      unsigned short* ko = out + ((long)(b * NUMH + h) * SS + s) * HD;
      ko[d0]      = f2bf(y0);
      ko[d0 + 16] = f2bf(y1);
    } else {
      int b = mOut / SS, s = mOut % SS;
      unsigned short* vt = out + (long)(b * NUMH + h) * HD * SPAD;
      vt[(long)d0 * SPAD + s]        = f2bf(y0);
      vt[(long)(d0 + 16) * SPAD + s] = f2bf(y1);
    }
  }
}

// ---------------------------------------------------------------- attention
// One 16-query tile per block (2 waves). scores->softmax->P@V all on-chip.
__global__ __launch_bounds__(64)
void attn_kernel(const unsigned short* __restrict__ Qb,   // (BH, KEEPN, 64)
                 const unsigned short* __restrict__ Kb,   // (BH, SS, 64)
                 const unsigned short* __restrict__ Vt,   // (BH, 64, SPAD)
                 float* __restrict__ out)                 // (BB, KEEPN, HID)
{
  __shared__ float          sS[16][SPAD];     // raw scores (f32)
  __shared__ unsigned short sP[16][SPAD];     // exp(s - max) (bf16), zero padded
  __shared__ float          sMax[16];
  __shared__ float          sScale[16];

  const int qblk = blockIdx.x;                // 0..18
  const int bh   = blockIdx.y;                // 0..383
  const int b    = bh / NUMH, h = bh % NUMH;
  const int tid  = threadIdx.x;
  const int lane = tid & 31, wave = tid >> 5;
  const int half = lane >> 4, nn = lane & 15;
  const int off  = half * 16;

  const unsigned short* Q  = Qb + (long)bh * KEEPN * HD;
  const unsigned short* K  = Kb + (long)bh * SS * HD;
  const unsigned short* VT = Vt + (long)bh * HD * SPAD;

  // Q A-fragments for K=0..31 and K=32..63 (row clamped for tail tile)
  const int qiC = imin(qblk * 16 + nn, KEEPN - 1);
  const unsigned short* qrow = Q + (long)qiC * HD;
  AFrag a0, a1;
  a0.h[0] = *(const v8bf*)(qrow + off);
  a0.h[1] = *(const v8bf*)(qrow + off + 8);
  a1.h[0] = *(const v8bf*)(qrow + 32 + off);
  a1.h[1] = *(const v8bf*)(qrow + 32 + off + 8);

  // ---- scores: Q (16x64) @ K_j^T (64x16), 37 key tiles split across 2 waves
  for (int j = wave; j < 37; j += 2) {
    const int sC = imin(j * 16 + nn, SS - 1);
    const unsigned short* krow = K + (long)sC * HD + half * 16;
    v16bf b0 = *(const v16bf*)(krow);
    v16bf b1 = *(const v16bf*)(krow + 32);
    v8f c = {};
    c = __builtin_amdgcn_wmma_f32_16x16x32_bf16(false, a0.v, false, b0, (short)0, c, false, false);
    c = __builtin_amdgcn_wmma_f32_16x16x32_bf16(false, a1.v, false, b1, (short)0, c, false, false);
#pragma unroll
    for (int r = 0; r < 8; ++r)
      sS[r + half * 8][j * 16 + nn] = c[r] * 0.125f;   // 1/sqrt(64)
  }
  __syncthreads();

  // ---- row max / sum (f32)
  if (tid < 16) {
    float mx = -INFINITY;
    for (int c = 0; c < SS; ++c) mx = fmaxf(mx, sS[tid][c]);
    float sum = 0.f;
    for (int c = 0; c < SS; ++c) sum += __expf(sS[tid][c] - mx);
    sMax[tid]   = mx;
    sScale[tid] = 1.0f / sum;
  }
  __syncthreads();

  // ---- probs (bf16, zero beyond 577)
  for (int r = 0; r < 16; ++r)
    for (int c = tid; c < SPAD; c += 64) {
      float p = (c < SS) ? __expf(sS[r][c] - sMax[r]) : 0.f;
      sP[r][c] = f2bf(p);
    }
  __syncthreads();

  // ---- ctx: P (16xSPAD) @ V (SPADx64); wave w covers d = w*32 .. w*32+31
  const int nt0 = wave * 2;
  const unsigned short* vrow0 = VT + (long)(nt0 * 16 + nn) * SPAD + half * 16;
  const unsigned short* vrow1 = VT + (long)(nt0 * 16 + 16 + nn) * SPAD + half * 16;
  v8f o0 = {}, o1 = {};
  for (int kk = 0; kk < SPAD; kk += 32) {
    AFrag a;
    a.h[0] = *(const v8bf*)(&sP[nn][kk + off]);
    a.h[1] = *(const v8bf*)(&sP[nn][kk + off + 8]);
    v16bf b0 = *(const v16bf*)(vrow0 + kk);
    v16bf b1 = *(const v16bf*)(vrow1 + kk);
    o0 = __builtin_amdgcn_wmma_f32_16x16x32_bf16(false, a.v, false, b0, (short)0, o0, false, false);
    o1 = __builtin_amdgcn_wmma_f32_16x16x32_bf16(false, a.v, false, b1, (short)0, o1, false, false);
  }

  const int d0 = nt0 * 16 + nn;
#pragma unroll
  for (int r = 0; r < 8; ++r) {
    int row = r + half * 8;
    int i = qblk * 16 + row;
    if (i >= KEEPN) continue;
    float scale = sScale[row];
    float* orow = out + (long)(b * KEEPN + i) * HID + h * HD;
    orow[d0]      = o0[r] * scale;
    orow[d0 + 16] = o1[r] * scale;
  }
}

// ---------------------------------------------------------------- launch
extern "C" void kernel_launch(void* const* d_in, const int* in_sizes, int n_in,
                              void* d_out, int out_size, void* d_ws, size_t ws_size,
                              hipStream_t stream) {
  const float* X   = (const float*)d_in[0];
  const float* Wq  = (const float*)d_in[1];
  const float* bq  = (const float*)d_in[2];
  const float* Wk  = (const float*)d_in[3];
  const float* bk  = (const float*)d_in[4];
  const float* Wv  = (const float*)d_in[5];
  const float* bv  = (const float*)d_in[6];
  const int*   idx = (const int*)d_in[7];
  float* out = (float*)d_out;

  char* ws = (char*)d_ws;
  size_t o = 0;
  auto alloc = [&](size_t bytes) {
    char* p = ws + o;
    o += (bytes + 255) & ~(size_t)255;
    return p;
  };
  unsigned short* Xb  = (unsigned short*)alloc((size_t)BB * SS * HID * 2);
  unsigned short* Wqb = (unsigned short*)alloc((size_t)HID * HID * 2);
  unsigned short* Wkb = (unsigned short*)alloc((size_t)HID * HID * 2);
  unsigned short* Wvb = (unsigned short*)alloc((size_t)HID * HID * 2);
  unsigned short* Qb  = (unsigned short*)alloc((size_t)BB * NUMH * KEEPN * HD * 2);
  unsigned short* Kb  = (unsigned short*)alloc((size_t)BB * NUMH * SS * HD * 2);
  unsigned short* Vtb = (unsigned short*)alloc((size_t)BB * NUMH * HD * SPAD * 2);

  const int nX = BB * SS * HID;
  const int nW = HID * HID;
  cvt_bf16_kernel<<<2048, 256, 0, stream>>>(X,  Xb,  nX);
  cvt_bf16_kernel<<<1024, 256, 0, stream>>>(Wq, Wqb, nW);
  cvt_bf16_kernel<<<1024, 256, 0, stream>>>(Wk, Wkb, nW);
  cvt_bf16_kernel<<<1024, 256, 0, stream>>>(Wv, Wvb, nW);

  const int Mq = BB * KEEPN;   // 9248
  const int Mk = BB * SS;      // 18464
  proj_kernel<<<dim3((Mq + 63) / 64, NUMH), 256, 0, stream>>>(Xb, Wqb, bq, idx, Qb, 0, Mq);
  proj_kernel<<<dim3((Mk + 63) / 64, NUMH), 256, 0, stream>>>(Xb, Wkb, bk, idx, Kb, 1, Mk);
  proj_kernel<<<dim3((Mk + 63) / 64, NUMH), 256, 0, stream>>>(Xb, Wvb, bv, idx, Vtb, 2, Mk);

  attn_kernel<<<dim3((KEEPN + 15) / 16, BB * NUMH), 64, 0, stream>>>(Qb, Kb, Vtb, out);
}
